// ViTBackbone_26371099197478
// MI455X (gfx1250) — matrix operand order repaired and checked by reading, not compiled, not run on hardware
//
#include <hip/hip_runtime.h>
#include <hip/hip_bf16.h>

#define DIM         512
#define HEADS       8
#define HEAD_DIM    64
#define DEPTH       4
#define MLP_DIM     1024
#define HPN         26
#define WPN         46
#define NUM_PATCHES 1196
#define WS2         100
#define N_PAD       1200
#define NW          12
#define BATCH       8
#define PATCH_DIM   3072
#define IMG_H       832
#define IMG_W       1472

typedef __attribute__((ext_vector_type(16))) _Float16 v16h;
typedef __attribute__((ext_vector_type(8)))  _Float16 v8h;
typedef __attribute__((ext_vector_type(8)))  float    v8f;

// ---- CDNA5 async global->LDS copy (16B per lane) + wait -------------------
__device__ __forceinline__ void async_b128(unsigned lds_off, unsigned goff,
                                           const void* sbase) {
  asm volatile("global_load_async_to_lds_b128 %0, %1, %2 offset:0"
               :: "v"(lds_off), "v"(goff), "s"(sbase) : "memory");
}
#define ASYNC_WAIT(n) asm volatile("s_wait_asynccnt " #n ::: "memory")
__device__ __forceinline__ unsigned lds_off_of(const void* p) {
  return (unsigned)(uintptr_t)p;  // generic LDS ptr: low 32 bits = LDS offset
}

// ---- WMMA fragment helpers ------------------------------------------------
// A frag (16x32 f16): lane<16 holds K{0..7,16..23}, lane>=16 K{8..15,24..31}
__device__ __forceinline__ v16h load_frag_a(const _Float16* rowp, int lane) {
  const _Float16* p = rowp + ((lane >= 16) ? 8 : 0);
  v8h lo = *(const v8h*)(p);
  v8h hi = *(const v8h*)(p + 16);
  v16h r;
#pragma unroll
  for (int i = 0; i < 8; ++i) { r[i] = lo[i]; r[i + 8] = hi[i]; }
  return r;
}
// B frag (32x16 f16) from Bt (N-major, K contiguous): lane<16 K{0..7,16..23}, else K{8..15,24..31}
__device__ __forceinline__ v16h load_frag_b(const _Float16* rowp, int lane) {
  const _Float16* p = rowp + ((lane >= 16) ? 8 : 0);
  v8h lo = *(const v8h*)(p);
  v8h hi = *(const v8h*)(p + 16);
  v16h r;
#pragma unroll
  for (int i = 0; i < 8; ++i) { r[i] = lo[i]; r[i + 8] = hi[i]; }
  return r;
}
__device__ __forceinline__ v8f wmma_f16(v16h a, v16h b, v8f c) {
  return __builtin_amdgcn_wmma_f32_16x16x32_f16(false, a, false, b, (short)0, c,
                                                false, false);
}
__device__ __forceinline__ float gelu_f(float x) {
  return 0.5f * x * (1.0f + erff(x * 0.70710678118654752f));
}
// patch_dim index k = (py*32+px)*3 + c ; value = img[b, c, hp*32+py, wp*32+px]
__device__ __forceinline__ float patch_elem(const float* __restrict__ img,
                                            int row, int k) {
  int b  = row / NUM_PATCHES;
  int p  = row - b * NUM_PATCHES;
  int hp = p / WPN;
  int wp = p - hp * WPN;
  int c  = k % 3;
  int t  = k / 3;
  int px = t & 31;
  int py = t >> 5;
  return img[((size_t)(b * 3 + c) * IMG_H + (size_t)(hp * 32 + py)) * IMG_W +
             (wp * 32 + px)];
}

// ---- Weight transpose: W (KxN f32) -> Wt (NxK f16, K contiguous) ----------
__global__ __launch_bounds__(256) void transpose_w(const float* __restrict__ W,
                                                   _Float16* __restrict__ Wt,
                                                   int K, int N) {
  int idx = blockIdx.x * 256 + threadIdx.x;
  if (idx >= K * N) return;
  int n = idx / K;
  int k = idx - n * K;
  Wt[idx] = (_Float16)W[(size_t)k * N + n];
}

// ---- WMMA GEMM: 128x128 block tile, BK=32, double-buffered async LDS ------
// AMODE: 0 = A is f16 row-major (async copy), 1 = fused patch gather (f32 img)
// EPI:   0 = bias, 1 = bias+GELU, 2 = bias + pos_emb[1+p]
// OUT:   0 = f32 store to Cf, 1 = f16 store to Ch
#define GP 48  // LDS row pitch in halves (96B, 16B-aligned chunks)
template <int AMODE, int EPI, int OUT>
__global__ __launch_bounds__(256) void gemm_wmma(
    const _Float16* __restrict__ A16, const _Float16* __restrict__ Bt,
    const float* __restrict__ bias, const float* __restrict__ pos,
    const float* __restrict__ img, float* __restrict__ Cf,
    _Float16* __restrict__ Ch, int M, int N, int K) {
  __shared__ __attribute__((aligned(16))) _Float16 sA[2][128 * GP];
  __shared__ __attribute__((aligned(16))) _Float16 sB[2][128 * GP];
  const int tid   = threadIdx.x;
  const int lane  = tid & 31;
  const int wave  = tid >> 5;
  const int wavem = wave & 1;   // 2 M-halves of 64
  const int waven = wave >> 1;  // 4 N-quarters of 32
  const int bM    = blockIdx.y * 128;
  const int bN    = blockIdx.x * 128;
  const int l15   = lane & 15;

  v8f acc[4][2];
  v8f z = {};
#pragma unroll
  for (int mt = 0; mt < 4; ++mt)
#pragma unroll
    for (int nt = 0; nt < 2; ++nt) acc[mt][nt] = z;

  const int rr = tid >> 1;   // 0..127 (tile row for copies)
  const int ch = tid & 1;    // 16B chunk within 32-half row

  // ---- stage issuer: copies tile (kb) into buffer s -----------------------
  auto issue = [&](int s, int kb) {
    if (AMODE == 0) {
      async_b128(lds_off_of(&sA[s][rr * GP + ch * 8]),
                 (unsigned)((((size_t)(bM + rr)) * K + kb + ch * 8) * 2), A16);
    } else {
#pragma unroll
      for (int j = 0; j < 16; ++j) {  // 128x32 f32 gather -> f16 LDS
        int idx = tid + j * 256;
        int r = idx >> 5, c = idx & 31;
        int row = bM + r;
        float v = (row < M) ? patch_elem(img, row, kb + c) : 0.0f;
        sA[s][r * GP + c] = (_Float16)v;
      }
    }
    async_b128(lds_off_of(&sB[s][rr * GP + ch * 8]),
               (unsigned)((((size_t)(bN + rr)) * K + kb + ch * 8) * 2), Bt);
  };

  const int nk = K >> 5;
  issue(0, 0);
  for (int t = 0; t < nk; ++t) {
    const int cur = t & 1;
    const bool hasnext = (t + 1 < nk);
    if (hasnext) issue(cur ^ 1, (t + 1) << 5);
    if (AMODE == 0) {
      if (hasnext) ASYNC_WAIT(0x2); else ASYNC_WAIT(0x0);
    } else {
      if (hasnext) ASYNC_WAIT(0x1); else ASYNC_WAIT(0x0);
    }
    __syncthreads();
    const _Float16* As = sA[cur];
    const _Float16* Bs = sB[cur];
    v16h bf[2];
#pragma unroll
    for (int nt = 0; nt < 2; ++nt)
      bf[nt] = load_frag_b(&Bs[(waven * 32 + nt * 16 + l15) * GP], lane);
#pragma unroll
    for (int mt = 0; mt < 4; ++mt) {
      v16h af = load_frag_a(&As[(wavem * 64 + mt * 16 + l15) * GP], lane);
#pragma unroll
      for (int nt = 0; nt < 2; ++nt)
        acc[mt][nt] = wmma_f16(af, bf[nt], acc[mt][nt]);
    }
    __syncthreads();
  }

  const int rofs = (lane >= 16) ? 8 : 0;  // C layout: vgpr r -> M=r / r+8
#pragma unroll
  for (int mt = 0; mt < 4; ++mt)
#pragma unroll
    for (int nt = 0; nt < 2; ++nt)
#pragma unroll
      for (int r = 0; r < 8; ++r) {
        int row = bM + wavem * 64 + mt * 16 + r + rofs;
        if (row >= M) continue;
        int col = bN + waven * 32 + nt * 16 + l15;
        float v = acc[mt][nt][r] + bias[col];
        if (EPI == 1) v = gelu_f(v);
        if (EPI == 2) {
          int p = row % NUM_PATCHES;
          v += pos[(size_t)(1 + p) * DIM + col];
        }
        if (OUT == 0) Cf[(size_t)row * N + col] = v;
        else          Ch[(size_t)row * N + col] = (_Float16)v;
      }
}

// ---- Fused window attention: one block per (batch, window, head) ----------
#define QK_PITCH 80    // 64 + 16 pad halves; 160B rows -> 16B-aligned chunks
#define PV_PITCH 136   // 128 + 8 pad (K padded 100 -> 128); 272B rows
__global__ __launch_bounds__(256) void window_attn(
    const _Float16* __restrict__ qkv16, _Float16* __restrict__ out16) {
  __shared__ __attribute__((aligned(16))) _Float16 s_qk[2 * 112 * QK_PITCH];
  __shared__ __attribute__((aligned(16))) _Float16 s_vt[64 * PV_PITCH];
  const int tid  = threadIdx.x;
  const int lane = tid & 31;
  const int wave = tid >> 5;
  const int bid  = blockIdx.x;
  const int b    = bid / (NW * HEADS);
  const int rem  = bid - b * (NW * HEADS);
  const int w    = rem >> 3;
  const int h    = rem & 7;
  const size_t rowbase = (size_t)(b * N_PAD + w * WS2) * (3 * DIM);  // halves
  _Float16* qs = s_qk;
  _Float16* ks = s_qk + 112 * QK_PITCH;
  _Float16* ps = s_qk;  // P overlays Q|K after barrier (15232 <= 17920 halves)
  const int l15 = lane & 15;

  // zero-fill pad rows m=100..111 of Q,K
  for (int idx = tid; idx < 12 * 64; idx += 256) {
    int m = 100 + (idx >> 6), d = idx & 63;
    qs[m * QK_PITCH + d] = (_Float16)0.0f;
    ks[m * QK_PITCH + d] = (_Float16)0.0f;
  }
  // async-copy Q,K rows 0..99 (8 x 16B chunks per 64-half row)
  for (int c = tid; c < 1600; c += 256) {
    int mat = (c >= 800) ? 1 : 0;
    int rc  = c - mat * 800;
    int m   = rc >> 3, cch = rc & 7;
    _Float16* dst = (mat ? ks : qs) + m * QK_PITCH + cch * 8;
    unsigned goff =
        (unsigned)((rowbase + (size_t)m * (3 * DIM) + mat * DIM + h * 64 +
                    cch * 8) * 2);
    async_b128(lds_off_of(dst), goff, qkv16);
  }
  // V^T staging (transpose, elementwise), zero pad kk>=100
  for (int idx = tid; idx < 64 * 128; idx += 256) {
    int d = idx & 63, kk = idx >> 6;
    _Float16 vv = (_Float16)0.0f;
    if (kk < WS2)
      vv = qkv16[rowbase + (size_t)kk * (3 * DIM) + 2 * DIM + h * 64 + d];
    s_vt[d * PV_PITCH + kk] = vv;
  }
  ASYNC_WAIT(0x0);
  __syncthreads();

  v8f acc[7];
  if (wave < 7) {  // S = Q K^T, one 16-row M-tile per wave, 7 N-tiles
    const int mt = wave;
    v8f z = {};
#pragma unroll
    for (int nt = 0; nt < 7; ++nt) acc[nt] = z;
#pragma unroll
    for (int kt = 0; kt < 2; ++kt) {
      v16h af = load_frag_a(&qs[(mt * 16 + l15) * QK_PITCH + kt * 32], lane);
#pragma unroll
      for (int nt = 0; nt < 7; ++nt) {
        v16h bf = load_frag_b(&ks[(nt * 16 + l15) * QK_PITCH + kt * 32], lane);
        acc[nt] = wmma_f16(af, bf, acc[nt]);
      }
    }
#pragma unroll
    for (int nt = 0; nt < 7; ++nt) {  // scale + mask cols >= 100
      bool msk = (nt == 6) && (l15 >= 4);
#pragma unroll
      for (int r = 0; r < 8; ++r)
        acc[nt][r] = msk ? -1e30f : acc[nt][r] * 0.125f;
    }
#pragma unroll
    for (int r = 0; r < 8; ++r) {  // row softmax within 16-lane groups
      float mx = acc[0][r];
#pragma unroll
      for (int nt = 1; nt < 7; ++nt) mx = fmaxf(mx, acc[nt][r]);
#pragma unroll
      for (int off = 1; off < 16; off <<= 1)
        mx = fmaxf(mx, __shfl_xor(mx, off, 32));
      float s = 0.0f;
#pragma unroll
      for (int nt = 0; nt < 7; ++nt) {
        float e = __expf(acc[nt][r] - mx);
        acc[nt][r] = e; s += e;
      }
#pragma unroll
      for (int off = 1; off < 16; off <<= 1) s += __shfl_xor(s, off, 32);
      float inv = 1.0f / s;
#pragma unroll
      for (int nt = 0; nt < 7; ++nt) acc[nt][r] *= inv;
    }
  }
  __syncthreads();  // all Q/K reads complete before P overlays them

  if (wave < 7) {  // spill P (f16) + zero pad cols 112..127
    const int mt = wave;
    const int rofs = (lane >= 16) ? 8 : 0;
#pragma unroll
    for (int nt = 0; nt < 7; ++nt)
#pragma unroll
      for (int r = 0; r < 8; ++r)
        ps[(mt * 16 + r + rofs) * PV_PITCH + nt * 16 + l15] =
            (_Float16)acc[nt][r];
    for (int i = lane; i < 256; i += 32) {
      int rw = i >> 4, cc = 112 + (i & 15);
      ps[(mt * 16 + rw) * PV_PITCH + cc] = (_Float16)0.0f;
    }
  }
  __syncthreads();

  if (wave < 7) {  // O = P V : 4 N-tiles x 4 K-steps
    const int mt = wave;
    v8f o[4];
    v8f z = {};
#pragma unroll
    for (int nt = 0; nt < 4; ++nt) o[nt] = z;
#pragma unroll
    for (int kt = 0; kt < 4; ++kt) {
      v16h af = load_frag_a(&ps[(mt * 16 + l15) * PV_PITCH + kt * 32], lane);
#pragma unroll
      for (int nt = 0; nt < 4; ++nt) {
        v16h bf = load_frag_b(&s_vt[(nt * 16 + l15) * PV_PITCH + kt * 32], lane);
        o[nt] = wmma_f16(af, bf, o[nt]);
      }
    }
    const int rofs = (lane >= 16) ? 8 : 0;
#pragma unroll
    for (int nt = 0; nt < 4; ++nt)
#pragma unroll
      for (int r = 0; r < 8; ++r) {
        int m = mt * 16 + r + rofs;
        if (m < WS2)
          out16[(size_t)(b * N_PAD + w * WS2 + m) * DIM + h * 64 + nt * 16 +
                l15] = (_Float16)o[nt][r];
      }
  }
}

// ---- LayerNorm (f32 in, f16 out): one wave per 512-wide row ---------------
__global__ __launch_bounds__(256) void layernorm_f16(
    const float* __restrict__ in, _Float16* __restrict__ out,
    const float* __restrict__ g, const float* __restrict__ bt, int n_in,
    int n_out, int n_valid, int nrows) {
  const int lane = threadIdx.x & 31;
  const int wave = threadIdx.x >> 5;
  const int rp   = blockIdx.x * 8 + wave;
  if (rp >= nrows) return;
  const int b = rp / n_out;
  const int p = rp - b * n_out;
  _Float16* orow = out + (size_t)rp * DIM;
  if (p >= n_valid) {  // zero-pad rows (reference zero-pads x before QKV)
#pragma unroll
    for (int i = 0; i < 16; ++i) orow[lane + i * 32] = (_Float16)0.0f;
    return;
  }
  const float* irow = in + (size_t)(b * n_in + p) * DIM;
  float x[16];
  float s = 0.0f, s2 = 0.0f;
#pragma unroll
  for (int i = 0; i < 16; ++i) {
    x[i] = irow[lane + i * 32];
    s += x[i]; s2 += x[i] * x[i];
  }
#pragma unroll
  for (int off = 1; off < 32; off <<= 1) {
    s  += __shfl_xor(s, off, 32);
    s2 += __shfl_xor(s2, off, 32);
  }
  float mean = s * (1.0f / DIM);
  float var  = s2 * (1.0f / DIM) - mean * mean;
  float rstd = rsqrtf(var + 1e-5f);
#pragma unroll
  for (int i = 0; i < 16; ++i) {
    int d = lane + i * 32;
    orow[d] = (_Float16)((x[i] - mean) * rstd * g[d] + bt[d]);
  }
}

// ---- Final assembly: out = concat(cls_tok + pos[0], pt) -------------------
__global__ __launch_bounds__(256) void assemble_k(const float* __restrict__ pt,
                                                  const float* __restrict__ cls_tok,
                                                  const float* __restrict__ pos,
                                                  float* __restrict__ out) {
  size_t idx = (size_t)blockIdx.x * 256 + threadIdx.x;
  const size_t per_b = (size_t)(NUM_PATCHES + 1) * DIM;
  size_t b = idx / per_b;
  size_t r = idx - b * per_b;
  int p = (int)(r / DIM);
  int d = (int)(r - (size_t)p * DIM);
  float v;
  if (p == 0) v = cls_tok[d] + pos[d];
  else        v = pt[((size_t)b * NUM_PATCHES + (p - 1)) * DIM + d];
  out[idx] = v;
}

extern "C" void kernel_launch(void* const* d_in, const int* in_sizes, int n_in,
                              void* d_out, int out_size, void* d_ws,
                              size_t ws_size, hipStream_t stream) {
  (void)in_sizes; (void)n_in; (void)out_size; (void)ws_size;
  const float* img     = (const float*)d_in[0];
  const float* patch_W = (const float*)d_in[1];
  const float* patch_b = (const float*)d_in[2];
  const float* pos_emb = (const float*)d_in[3];
  const float* cls_tok = (const float*)d_in[4];
  const float* ln1_g   = (const float*)d_in[5];
  const float* ln1_b   = (const float*)d_in[6];
  const float* qkv_W   = (const float*)d_in[7];
  const float* qkv_b   = (const float*)d_in[8];
  const float* proj_W  = (const float*)d_in[9];
  const float* proj_b  = (const float*)d_in[10];
  const float* ln2_g   = (const float*)d_in[11];
  const float* ln2_b   = (const float*)d_in[12];
  const float* mlp_W1  = (const float*)d_in[13];
  const float* mlp_b1  = (const float*)d_in[14];
  const float* mlp_W2  = (const float*)d_in[15];
  const float* mlp_b2  = (const float*)d_in[16];
  float* out = (float*)d_out;

  // ---- workspace layout (all bases 16B-aligned) ---------------------------
  char* wsp = (char*)d_ws;
  float* pt = (float*)wsp;     wsp += (size_t)BATCH * NUM_PATCHES * DIM * 4;
  float* pj = (float*)wsp;     wsp += (size_t)BATCH * N_PAD * DIM * 4;
  _Float16* x16   = (_Float16*)wsp; wsp += (size_t)BATCH * N_PAD * DIM * 2;
  _Float16* qkv16 = (_Float16*)wsp; wsp += (size_t)BATCH * N_PAD * 3 * DIM * 2;
  _Float16* h16   = (_Float16*)wsp; wsp += (size_t)BATCH * N_PAD * DIM * 2;
  _Float16* mh16  = (_Float16*)wsp; wsp += (size_t)BATCH * N_PAD * MLP_DIM * 2;
  _Float16* wt_patch = (_Float16*)wsp; wsp += (size_t)DIM * PATCH_DIM * 2;
  _Float16* wt_qkv   = (_Float16*)wsp; wsp += (size_t)DEPTH * 3 * DIM * DIM * 2;
  _Float16* wt_proj  = (_Float16*)wsp; wsp += (size_t)DEPTH * DIM * DIM * 2;
  _Float16* wt_m1    = (_Float16*)wsp; wsp += (size_t)DEPTH * MLP_DIM * DIM * 2;
  _Float16* wt_m2    = (_Float16*)wsp; wsp += (size_t)DEPTH * DIM * MLP_DIM * 2;

  const int Mp = BATCH * NUM_PATCHES;  // 9568
  const int Mx = BATCH * N_PAD;        // 9600
  dim3 blk(256);

  // ---- one-shot weight transposes to f16 N-major --------------------------
  transpose_w<<<dim3((PATCH_DIM * DIM + 255) / 256), blk, 0, stream>>>(
      patch_W, wt_patch, PATCH_DIM, DIM);
  for (int i = 0; i < DEPTH; ++i) {
    transpose_w<<<dim3((DIM * 3 * DIM + 255) / 256), blk, 0, stream>>>(
        qkv_W + (size_t)i * DIM * 3 * DIM, wt_qkv + (size_t)i * 3 * DIM * DIM,
        DIM, 3 * DIM);
    transpose_w<<<dim3((DIM * DIM + 255) / 256), blk, 0, stream>>>(
        proj_W + (size_t)i * DIM * DIM, wt_proj + (size_t)i * DIM * DIM, DIM,
        DIM);
    transpose_w<<<dim3((DIM * MLP_DIM + 255) / 256), blk, 0, stream>>>(
        mlp_W1 + (size_t)i * DIM * MLP_DIM, wt_m1 + (size_t)i * MLP_DIM * DIM,
        DIM, MLP_DIM);
    transpose_w<<<dim3((MLP_DIM * DIM + 255) / 256), blk, 0, stream>>>(
        mlp_W2 + (size_t)i * MLP_DIM * DIM, wt_m2 + (size_t)i * DIM * MLP_DIM,
        MLP_DIM, DIM);
  }

  // patch embedding (fused im2col) + bias + positional embedding -> pt (f32)
  gemm_wmma<1, 2, 0><<<dim3(DIM / 128, (Mp + 127) / 128), blk, 0, stream>>>(
      nullptr, wt_patch, patch_b, pos_emb, img, pt, nullptr, Mp, DIM,
      PATCH_DIM);

  for (int i = 0; i < DEPTH; ++i) {
    layernorm_f16<<<dim3(Mx / 8), blk, 0, stream>>>(
        pt, x16, ln1_g + i * DIM, ln1_b + i * DIM, NUM_PATCHES, N_PAD,
        NUM_PATCHES, Mx);
    gemm_wmma<0, 0, 1><<<dim3(3 * DIM / 128, Mx / 128), blk, 0, stream>>>(
        x16, wt_qkv + (size_t)i * 3 * DIM * DIM, qkv_b + i * 3 * DIM, nullptr,
        nullptr, nullptr, qkv16, Mx, 3 * DIM, DIM);
    window_attn<<<dim3(BATCH * NW * HEADS), blk, 0, stream>>>(qkv16, x16);
    gemm_wmma<0, 0, 0><<<dim3(DIM / 128, Mx / 128), blk, 0, stream>>>(
        x16, wt_proj + (size_t)i * DIM * DIM, proj_b + i * DIM, nullptr,
        nullptr, pj, nullptr, Mx, DIM, DIM);
    layernorm_f16<<<dim3(Mp / 8), blk, 0, stream>>>(
        pj, h16, ln2_g + i * DIM, ln2_b + i * DIM, N_PAD, NUM_PATCHES,
        NUM_PATCHES, Mp);
    gemm_wmma<0, 1, 1><<<dim3(MLP_DIM / 128, (Mp + 127) / 128), blk, 0,
                         stream>>>(h16, wt_m1 + (size_t)i * MLP_DIM * DIM,
                                   mlp_b1 + i * MLP_DIM, nullptr, nullptr,
                                   nullptr, mh16, Mp, MLP_DIM, DIM);
    gemm_wmma<0, 0, 0><<<dim3(DIM / 128, (Mp + 127) / 128), blk, 0, stream>>>(
        mh16, wt_m2 + (size_t)i * DIM * MLP_DIM, mlp_b2 + i * DIM, nullptr,
        nullptr, pt, nullptr, Mp, DIM, MLP_DIM);
  }

  size_t tot = (size_t)BATCH * (NUM_PATCHES + 1) * DIM;  // 4,902,912
  assemble_k<<<dim3((unsigned)(tot / 256)), blk, 0, stream>>>(pt, cls_tok,
                                                              pos_emb, out);
}
